// RelativeVisionTransformerSPP_90726889160765
// MI455X (gfx1250) — compile-verified, matrix operand order checked by reading, not run on hardware
//
#include <hip/hip_runtime.h>

// ---------------- constants ----------------
#define BB 8
#define HWD 224
#define NPIX (BB*HWD*HWD)      // 401408
#define DIMC 256
#define DHD 64
#define NHD 8
#define MLPD 1024
#define NLD 6
#define NOUTD 1000
#define TT 86
#define EPSF 1e-5f
#define STATS_NB 512

typedef __bf16 v16bf __attribute__((ext_vector_type(16)));
typedef float  v8f   __attribute__((ext_vector_type(8)));

__device__ __forceinline__ unsigned short f2bf(float f) {
    unsigned u = __builtin_bit_cast(unsigned, f);
    unsigned r = u + 0x7fffu + ((u >> 16) & 1u);   // round-to-nearest-even
    return (unsigned short)(r >> 16);
}
__device__ __forceinline__ float bf2f(unsigned short h) {
    unsigned u = ((unsigned)h) << 16;
    return __builtin_bit_cast(float, u);
}

// LDS tile rows are padded to 40 ushorts (80B) -> conflict-free ds_read_b128 pairs
#define LROW 40

// Load one 16x32 bf16 WMMA operand fragment from an LDS tile ([row][k], padded).
// lane 0-15: rows r0..r0+15, K halves {0..7,16..23}; lane 16-31: same rows, K {8..15,24..31}.
__device__ __forceinline__ v16bf frag_load(const unsigned short* base, int row0, int lane) {
    const unsigned short* p = base + (row0 + (lane & 15)) * LROW;
    int k0 = (lane & 16) ? 8 : 0;
    union { uint4 q[2]; v16bf v; } u;
    u.q[0] = *reinterpret_cast<const uint4*>(p + k0);
    u.q[1] = *reinterpret_cast<const uint4*>(p + k0 + 16);
    return u.v;
}

// ================= conv1: direct 3x3, Cin=3 -> Cout=32, f32 math, bf16 NHWC out ====
__global__ __launch_bounds__(256) void k_conv1(const float* __restrict__ img,
                                               const float* __restrict__ w,
                                               unsigned short* __restrict__ out) {
    __shared__ float sw[32 * 27];
    int tid = threadIdx.x;
    for (int i = tid; i < 864; i += 256) sw[i] = w[i];
    __syncthreads();
    long gid = (long)blockIdx.x * 256 + tid;
    int c = (int)(gid & 31);
    long p = gid >> 5;
    int b = (int)(p / (HWD * HWD));
    int rem = (int)(p - (long)b * HWD * HWD);
    int y = rem / HWD, x = rem - (rem / HWD) * HWD;
    float acc = 0.f;
    for (int ci = 0; ci < 3; ++ci) {
        const float* ip = img + ((long)(b * 3 + ci) * HWD) * HWD;
        for (int ky = 0; ky < 3; ++ky) {
            int yy = y + ky - 1;
            if (yy < 0 || yy >= HWD) continue;
            for (int kx = 0; kx < 3; ++kx) {
                int xx = x + kx - 1;
                if (xx < 0 || xx >= HWD) continue;
                acc += ip[yy * HWD + xx] * sw[c * 27 + ci * 9 + ky * 3 + kx];
            }
        }
    }
    out[gid] = f2bf(acc);
}

// ============ per-channel BN stats, deterministic 2-stage (no atomics) ============
__global__ __launch_bounds__(256) void k_chanstats(const unsigned short* __restrict__ x,
                                                   float* __restrict__ psum,
                                                   float* __restrict__ psq,
                                                   int C, int npix) {
    int t = threadIdx.x;
    int c = t % C;
    int rowsPer = 256 / C;
    int sub = t / C;
    long stride = (long)gridDim.x * rowsPer;
    float s = 0.f, s2 = 0.f;
    for (long p = (long)blockIdx.x * rowsPer + sub; p < npix; p += stride) {
        float v = bf2f(x[p * C + c]);
        s += v; s2 += v * v;
    }
    __shared__ float ls[256], ls2[256];
    ls[t] = s; ls2[t] = s2;
    __syncthreads();
    if (t < C) {
        float a = 0.f, a2 = 0.f;
        for (int k = 0; k < 256 / C; ++k) { a += ls[k * C + t]; a2 += ls2[k * C + t]; }
        psum[blockIdx.x * C + t] = a;
        psq [blockIdx.x * C + t] = a2;
    }
}

__global__ void k_bnfin(const float* __restrict__ psum, const float* __restrict__ psq,
                        const float* __restrict__ g, const float* __restrict__ beta,
                        float* __restrict__ aff_a, float* __restrict__ aff_b,
                        int C, float inv_n) {
    int c = blockIdx.x * blockDim.x + threadIdx.x;
    if (c >= C) return;
    float s = 0.f, s2 = 0.f;
    for (int nb = 0; nb < STATS_NB; ++nb) { s += psum[nb * C + c]; s2 += psq[nb * C + c]; }
    float mu = s * inv_n;
    float var = s2 * inv_n - mu * mu;
    float a = g[c] * rsqrtf(var + EPSF);
    aff_a[c] = a;
    aff_b[c] = beta[c] - mu * a;
}

__global__ __launch_bounds__(256) void k_bnrelu(const unsigned short* __restrict__ in,
                                                const float* __restrict__ aff_a,
                                                const float* __restrict__ aff_b,
                                                unsigned short* __restrict__ out,
                                                int C, long n) {
    long gid = (long)blockIdx.x * 256 + threadIdx.x;
    if (gid >= n) return;
    int c = (int)(gid % C);
    float v = bf2f(in[gid]);
    v = fmaxf(0.f, aff_a[c] * v + aff_b[c]);
    out[gid] = f2bf(v);
}

// ======== implicit-GEMM conv 3x3 pad1 via WMMA bf16, ping-pong LDS pipeline ========
// X: bf16 NHWC [B,224,224,CIN]; Wt: bf16 [COUT][9*CIN]; out: bf16 NHWC [NPIX][COUT]
// Block tile: 128 pixels x NT channels; 8 waves; wave = 32 chan x (NT/2) pixels.
template <int CIN, int COUT, int NT>
__global__ __launch_bounds__(256) void k_conv_igemm(const unsigned short* __restrict__ X,
                                                    const unsigned short* __restrict__ Wt,
                                                    unsigned short* __restrict__ out) {
    constexpr int KTOT  = 9 * CIN;
    constexpr int NITER = KTOT / 32;
    constexpr int NW_N  = NT / 32;       // waves along channel dim
    constexpr int NW_M  = 8 / NW_N;      // waves along pixel dim
    constexpr int MT    = 128 / NW_M;    // pixels per wave
    constexpr int TJ    = MT / 16;       // 16-wide pixel tiles per wave
    constexpr int ROWS  = 128 + NT;
    constexpr int WL    = NT * 4 / 256;  // weight uint4 loads per thread (1 or 2)

    __shared__ unsigned short lds[2 * ROWS * LROW];

    int tid = threadIdx.x, lane = tid & 31, wave = tid >> 5;
    int wn = wave % NW_N, wm = wave / NW_N;
    int m0 = blockIdx.x * 128, n0 = blockIdx.y * NT;

    int srow = tid >> 1, spart = tid & 1;          // pixel staging role
    int p = m0 + srow;
    int b = p / (HWD * HWD);
    int rem = p - b * HWD * HWD;
    int py = rem / HWD, px = rem - (rem / HWD) * HWD;

    uint4 pq0, pq1;          // staged pixel data
    uint4 wq[WL];            // staged weight data

    auto load_tile = [&](int it) {
        const int k0 = it * 32;
        const int kk = k0 / CIN;             // compile-time after unroll
        const int cb = k0 - kk * CIN;
        const int dy = kk / 3 - 1, dx = kk % 3 - 1;
        int yy = py + dy, xx = px + dx;
        pq0 = (uint4){0, 0, 0, 0}; pq1 = (uint4){0, 0, 0, 0};
        if (yy >= 0 && yy < HWD && xx >= 0 && xx < HWD) {
            const uint4* sp = reinterpret_cast<const uint4*>(
                X + ((long)((b * HWD + yy) * HWD + xx)) * CIN + cb + spart * 16);
            pq0 = sp[0]; pq1 = sp[1];
        }
        #pragma unroll
        for (int w = 0; w < WL; ++w) {
            int idx = tid + w * 256;
            int row = idx >> 2, q = idx & 3;
            wq[w] = *reinterpret_cast<const uint4*>(
                Wt + (long)(n0 + row) * KTOT + k0 + q * 8);
        }
    };
    auto store_tile = [&](int buf) {
        unsigned short* sPix = lds + buf * ROWS * LROW;
        unsigned short* sW   = sPix + 128 * LROW;
        *reinterpret_cast<uint4*>(sPix + srow * LROW + spart * 16)     = pq0;
        *reinterpret_cast<uint4*>(sPix + srow * LROW + spart * 16 + 8) = pq1;
        #pragma unroll
        for (int w = 0; w < WL; ++w) {
            int idx = tid + w * 256;
            int row = idx >> 2, q = idx & 3;
            *reinterpret_cast<uint4*>(sW + row * LROW + q * 8) = wq[w];
        }
    };

    v8f acc[2][TJ] = {};
    load_tile(0);
    store_tile(0);
    #pragma unroll
    for (int it = 0; it < NITER; ++it) {
        __syncthreads();                       // buf[it&1] ready; buf[(it+1)&1] free
        if (it + 1 < NITER) load_tile(it + 1); // global loads overlap WMMA below
        const unsigned short* sPix = lds + (it & 1) * ROWS * LROW;
        const unsigned short* sW   = sPix + 128 * LROW;
        v16bf af0 = frag_load(sW, wn * 32, lane);
        v16bf af1 = frag_load(sW, wn * 32 + 16, lane);
        v16bf bfr[TJ];
        #pragma unroll
        for (int j = 0; j < TJ; ++j) bfr[j] = frag_load(sPix, wm * MT + j * 16, lane);
        #pragma unroll
        for (int j = 0; j < TJ; ++j) {
            acc[0][j] = __builtin_amdgcn_wmma_f32_16x16x32_bf16(false, af0, false, bfr[j], (short)0, acc[0][j], false, false);
            acc[1][j] = __builtin_amdgcn_wmma_f32_16x16x32_bf16(false, af1, false, bfr[j], (short)0, acc[1][j], false, false);
        }
        if (it + 1 < NITER) store_tile((it + 1) & 1);
    }
    // epilogue: lane = pixel, VGPRs = 8 consecutive channels -> packed 16B store
    #pragma unroll
    for (int i = 0; i < 2; ++i)
        #pragma unroll
        for (int j = 0; j < TJ; ++j) {
            int pix = m0 + wm * MT + j * 16 + (lane & 15);
            int col = n0 + wn * 32 + i * 16 + ((lane & 16) ? 8 : 0);
            union { unsigned short h[8]; uint4 q; } u;
            #pragma unroll
            for (int r = 0; r < 8; ++r) u.h[r] = f2bf(acc[i][j][r]);
            *reinterpret_cast<uint4*>(out + (long)pix * COUT + col) = u.q;
        }
}

// ===== generic WMMA GEMM: out[M][N] = A[M][K](f32) * Wt[N][K](bf16)^T + bias + res ==
__global__ __launch_bounds__(256) void k_gemm(const float* __restrict__ A,
                                              const unsigned short* __restrict__ Wt,
                                              const float* __restrict__ bias,
                                              const float* __restrict__ res,
                                              float* __restrict__ out,
                                              int M, int N, int K) {
    __shared__ unsigned short lds[192 * LROW];
    unsigned short* sIn = lds;               // rows 0..127 : A rows (WMMA B operand)
    unsigned short* sW  = lds + 128 * LROW;  // rows 0..63  : Wt rows (WMMA A operand)
    int tid = threadIdx.x, lane = tid & 31, wave = tid >> 5;
    int wn = wave & 1, wm = wave >> 1;
    int m0 = blockIdx.x * 128, n0 = blockIdx.y * 64;
    int srow = tid >> 1, spart = tid & 1;
    int wrow = tid >> 2, wq = tid & 3;

    v8f acc[2][2] = {};
    for (int k0 = 0; k0 < K; k0 += 32) {
        { // stage 128 A rows, f32 -> bf16
            int m = m0 + srow;
            union { unsigned short h[16]; uint4 q[2]; } u;
            if (m < M) {
                const float4* ap = reinterpret_cast<const float4*>(
                    A + (long)m * K + k0 + spart * 16);
                #pragma unroll
                for (int v = 0; v < 4; ++v) {
                    float4 f = ap[v];
                    u.h[v * 4 + 0] = f2bf(f.x); u.h[v * 4 + 1] = f2bf(f.y);
                    u.h[v * 4 + 2] = f2bf(f.z); u.h[v * 4 + 3] = f2bf(f.w);
                }
            } else {
                u.q[0] = (uint4){0, 0, 0, 0}; u.q[1] = (uint4){0, 0, 0, 0};
            }
            *reinterpret_cast<uint4*>(sIn + srow * LROW + spart * 16)     = u.q[0];
            *reinterpret_cast<uint4*>(sIn + srow * LROW + spart * 16 + 8) = u.q[1];
        }
        { // stage 64 weight rows
            const uint4* wp = reinterpret_cast<const uint4*>(
                Wt + (long)(n0 + wrow) * K + k0 + wq * 8);
            *reinterpret_cast<uint4*>(sW + wrow * LROW + wq * 8) = *wp;
        }
        __syncthreads();
        v16bf af0 = frag_load(sW, wn * 32, lane);
        v16bf af1 = frag_load(sW, wn * 32 + 16, lane);
        v16bf bf0 = frag_load(sIn, wm * 32, lane);
        v16bf bf1 = frag_load(sIn, wm * 32 + 16, lane);
        acc[0][0] = __builtin_amdgcn_wmma_f32_16x16x32_bf16(false, af0, false, bf0, (short)0, acc[0][0], false, false);
        acc[0][1] = __builtin_amdgcn_wmma_f32_16x16x32_bf16(false, af0, false, bf1, (short)0, acc[0][1], false, false);
        acc[1][0] = __builtin_amdgcn_wmma_f32_16x16x32_bf16(false, af1, false, bf0, (short)0, acc[1][0], false, false);
        acc[1][1] = __builtin_amdgcn_wmma_f32_16x16x32_bf16(false, af1, false, bf1, (short)0, acc[1][1], false, false);
        __syncthreads();
    }
    #pragma unroll
    for (int i = 0; i < 2; ++i)
        #pragma unroll
        for (int j = 0; j < 2; ++j) {
            int row = m0 + wm * 32 + j * 16 + (lane & 15);
            int col = n0 + wn * 32 + i * 16 + ((lane & 16) ? 8 : 0);
            if (row < M) {
                long off = (long)row * N + col;
                float v[8];
                #pragma unroll
                for (int r = 0; r < 8; ++r) v[r] = acc[i][j][r];
                if (bias) {
                    #pragma unroll
                    for (int r = 0; r < 8; ++r) v[r] += bias[col + r];
                }
                if (res) {
                    #pragma unroll
                    for (int r = 0; r < 8; ++r) v[r] += res[off + r];
                }
                float4* o4 = reinterpret_cast<float4*>(out + off);
                o4[0] = make_float4(v[0], v[1], v[2], v[3]);
                o4[1] = make_float4(v[4], v[5], v[6], v[7]);
            }
        }
}

// ======= fused BN-affine + ReLU + level-8 adaptive pooling (28x28 regions) =========
__global__ __launch_bounds__(256) void k_pool8(const unsigned short* __restrict__ c3,
                                               const float* __restrict__ aff_a,
                                               const float* __restrict__ aff_b,
                                               float* __restrict__ pool8) {
    int c = threadIdx.x;
    int bx = blockIdx.x % 8, by = (blockIdx.x / 8) % 8, b = blockIdx.x / 64;
    float a = aff_a[c], bb = aff_b[c], s = 0.f;
    for (int y = by * 28; y < by * 28 + 28; ++y)
        for (int x = bx * 28; x < bx * 28 + 28; ++x) {
            float v = bf2f(c3[(((long)b * HWD + y) * HWD + x) * DIMC + c]);
            s += fmaxf(0.f, a * v + bb);
        }
    pool8[(long)blockIdx.x * DIMC + c] = s * (1.f / 784.f);
}

// levels 1/2/4 are exact averages of nested level-8 bins; token 0 = cls
__global__ void k_tokens(const float* __restrict__ pool8, const float* __restrict__ cls,
                         float* __restrict__ x) {
    int gid = blockIdx.x * 256 + threadIdx.x;
    if (gid >= BB * TT * DIMC) return;
    int c = gid % DIMC;
    int t = (gid / DIMC) % TT;
    int b = gid / (DIMC * TT);
    float v;
    if (t == 0) v = cls[c];
    else {
        int tt = t - 1, lvl, base;
        if (tt < 1)       { lvl = 1; base = 0; }
        else if (tt < 5)  { lvl = 2; base = 1; }
        else if (tt < 21) { lvl = 4; base = 5; }
        else              { lvl = 8; base = 21; }
        int r = tt - base, i = r / lvl, j = r % lvl, f = 8 / lvl;
        float s = 0.f;
        for (int fy = 0; fy < f; ++fy)
            for (int fx = 0; fx < f; ++fx)
                s += pool8[(((long)b * 8 + i * f + fy) * 8 + j * f + fx) * DIMC + c];
        v = s / (float)(f * f);
    }
    x[gid] = v;
}

// ================================ LayerNorm (D=256) ================================
__global__ __launch_bounds__(256) void k_ln(const float* __restrict__ x,
                                            const float* __restrict__ g,
                                            const float* __restrict__ beta,
                                            float* __restrict__ out,
                                            int rs_in, int rs_out) {
    __shared__ float red[256];
    int r = blockIdx.x, t = threadIdx.x;
    float v = x[(long)r * rs_in + t];
    red[t] = v; __syncthreads();
    for (int s = 128; s > 0; s >>= 1) { if (t < s) red[t] += red[t + s]; __syncthreads(); }
    float mu = red[0] * (1.f / 256.f);
    __syncthreads();
    float d = v - mu;
    red[t] = d * d; __syncthreads();
    for (int s = 128; s > 0; s >>= 1) { if (t < s) red[t] += red[t + s]; __syncthreads(); }
    float var = red[0] * (1.f / 256.f);
    out[(long)r * rs_out + t] = d * rsqrtf(var + EPSF) * g[t] + beta[t];
}

// =========================== Q·rel^T  (pre-skew table) =============================
__global__ void k_qrel(const float* __restrict__ qkv, const float* __restrict__ rel,
                       float* __restrict__ qrel) {
    long gid = (long)blockIdx.x * 256 + threadIdx.x;
    if (gid >= (long)BB * NHD * TT * TT) return;
    int l = (int)(gid % TT);
    int i = (int)((gid / TT) % TT);
    int h = (int)((gid / (TT * TT)) % NHD);
    int b = (int)(gid / ((long)TT * TT * NHD));
    const float* qp = qkv + ((long)(b * TT + i) * (3 * NHD * DHD)) + h * DHD;   // proj 0
    const float* rp = rel + ((long)h * TT + l) * DHD;
    float s = 0.f;
    #pragma unroll
    for (int k = 0; k < DHD; ++k) s += qp[k] * rp[k];
    qrel[gid] = s;
}

// ============ attention: scores + exact Music-Transformer skew + softmax ===========
__global__ __launch_bounds__(128) void k_attn(const float* __restrict__ qkv,
                                              const float* __restrict__ qrel,
                                              float* __restrict__ o) {
    int t = blockIdx.x % TT;
    int h = (blockIdx.x / TT) % NHD;
    int b = blockIdx.x / (TT * NHD);
    __shared__ float q[DHD];
    __shared__ float sc[TT];
    __shared__ float sinv;
    int tid = threadIdx.x;
    const int RS = 3 * NHD * DHD;   // 1536
    if (tid < DHD) q[tid] = qkv[((long)(b * TT + t) * RS) + h * DHD + tid];
    __syncthreads();
    if (tid < TT) {
        const float* kp = qkv + ((long)(b * TT + tid) * RS) + NHD * DHD + h * DHD;
        float s = 0.f;
        #pragma unroll
        for (int k = 0; k < DHD; ++k) s += q[k] * kp[k];
        // exact skew remap: f=(t+1)*T+j ; (i2,j2)=(f/(T+1), f%(T+1)); 0 if j2==0
        int f = (t + 1) * TT + tid;
        int i2 = f / (TT + 1);
        int j2 = f - i2 * (TT + 1);
        float r = (j2 == 0) ? 0.f
                : qrel[(((long)(b * NHD + h) * TT + i2) * TT) + (j2 - 1)];
        sc[tid] = (s + r) * 0.125f;   // / sqrt(64)
    }
    __syncthreads();
    if (tid == 0) {
        float m = -1e30f;
        for (int s = 0; s < TT; ++s) m = fmaxf(m, sc[s]);
        float sum = 0.f;
        for (int s = 0; s < TT; ++s) { float e = expf(sc[s] - m); sc[s] = e; sum += e; }
        sinv = 1.f / sum;
    }
    __syncthreads();
    if (tid < DHD) {
        float acc = 0.f;
        for (int s = 0; s < TT; ++s) {
            const float* vp = qkv + ((long)(b * TT + s) * RS) + 2 * NHD * DHD + h * DHD;
            acc += sc[s] * vp[tid];
        }
        o[((long)(b * TT + t) * (NHD * DHD)) + h * DHD + tid] = acc * sinv;
    }
}

// ================================= GELU (exact) ====================================
__global__ void k_gelu(float* __restrict__ x, long n) {
    long gid = (long)blockIdx.x * 256 + threadIdx.x;
    if (gid >= n) return;
    float v = x[gid];
    x[gid] = 0.5f * v * (1.f + erff(v * 0.70710678118654752f));
}

// ================================ head (cls only) ==================================
__global__ void k_head(const float* __restrict__ hrow, const float* __restrict__ Wh,
                       const float* __restrict__ bh, float* __restrict__ out) {
    int gid = blockIdx.x * 256 + threadIdx.x;
    if (gid >= BB * NOUTD) return;
    int n = gid % NOUTD, b = gid / NOUTD;
    float s = bh[n];
    const float* hr = hrow + (long)b * DIMC;
    for (int d = 0; d < DIMC; ++d) s += hr[d] * Wh[(long)d * NOUTD + n];
    out[gid] = s;
}

// ============================ weight conversion kernels ============================
// OIHW f32 -> [O][(ky*3+kx)*I + i] bf16
__global__ void k_cvt_convw(const float* __restrict__ src, unsigned short* __restrict__ dst,
                            int O, int I) {
    long gid = (long)blockIdx.x * 256 + threadIdx.x;
    long total = (long)O * 9 * I;
    if (gid >= total) return;
    int o = (int)(gid / (9 * I));
    int r = (int)(gid % (9 * I));
    int kk = r / I, i = r % I;
    dst[gid] = f2bf(src[((long)(o * I + i)) * 9 + kk]);
}
// Wq/Wk/Wv [L][H][D][K] f32 -> fused [L][col=proj*512+h*64+k][D] bf16
__global__ void k_cvt_qkvw(const float* __restrict__ Wq, const float* __restrict__ Wk,
                           const float* __restrict__ Wv, unsigned short* __restrict__ dst) {
    long gid = (long)blockIdx.x * 256 + threadIdx.x;
    long total = (long)NLD * 1536 * DIMC;
    if (gid >= total) return;
    int d = (int)(gid % DIMC);
    int col = (int)((gid / DIMC) % 1536);
    int l = (int)(gid / (DIMC * 1536));
    int proj = col / 512, h = (col % 512) / DHD, k = col % DHD;
    const float* W = (proj == 0) ? Wq : (proj == 1) ? Wk : Wv;
    dst[gid] = f2bf(W[(((long)(l * NHD + h) * DIMC + d)) * DHD + k]);
}
__global__ void k_cvt_qkvb(const float* __restrict__ bq, const float* __restrict__ bk,
                           const float* __restrict__ bv, float* __restrict__ dst) {
    int gid = blockIdx.x * 256 + threadIdx.x;
    if (gid >= NLD * 1536) return;
    int col = gid % 1536, l = gid / 1536;
    int proj = col / 512, h = (col % 512) / DHD, k = col % DHD;
    const float* bsrc = (proj == 0) ? bq : (proj == 1) ? bk : bv;
    dst[gid] = bsrc[(l * NHD + h) * DHD + k];
}
// generic [L][K][N] f32 -> [L][N][K] bf16
__global__ void k_cvt_t(const float* __restrict__ src, unsigned short* __restrict__ dst,
                        int L, int K, int N) {
    long gid = (long)blockIdx.x * 256 + threadIdx.x;
    long total = (long)L * K * N;
    if (gid >= total) return;
    int k = (int)(gid % K);
    int n = (int)((gid / K) % N);
    int l = (int)(gid / ((long)K * N));
    dst[gid] = f2bf(src[((long)l * K + k) * N + n]);
}

// ====================================================================================
extern "C" void kernel_launch(void* const* d_in, const int* in_sizes, int n_in,
                              void* d_out, int out_size, void* d_ws, size_t ws_size,
                              hipStream_t stream) {
    const float* img     = (const float*)d_in[0];
    const float* conv1_w = (const float*)d_in[1];
    const float* bn1_g   = (const float*)d_in[3];
    const float* bn1_b   = (const float*)d_in[4];
    const float* conv2_w = (const float*)d_in[5];
    const float* bn2_g   = (const float*)d_in[7];
    const float* bn2_b   = (const float*)d_in[8];
    const float* conv3_w = (const float*)d_in[9];
    const float* bn3_g   = (const float*)d_in[11];
    const float* bn3_b   = (const float*)d_in[12];
    const float* cls     = (const float*)d_in[13];
    const float* Wq      = (const float*)d_in[14];
    const float* bq      = (const float*)d_in[15];
    const float* Wk      = (const float*)d_in[16];
    const float* bk      = (const float*)d_in[17];
    const float* Wv      = (const float*)d_in[18];
    const float* bv      = (const float*)d_in[19];
    const float* rel     = (const float*)d_in[20];
    const float* Wo      = (const float*)d_in[21];
    const float* bo      = (const float*)d_in[22];
    const float* ln_g    = (const float*)d_in[23];
    const float* ln_b    = (const float*)d_in[24];
    const float* W1      = (const float*)d_in[25];
    const float* b1      = (const float*)d_in[26];
    const float* W2      = (const float*)d_in[27];
    const float* b2      = (const float*)d_in[28];
    const float* hln_g   = (const float*)d_in[29];
    const float* hln_b   = (const float*)d_in[30];
    const float* Wh      = (const float*)d_in[31];
    const float* bh      = (const float*)d_in[32];
    float* outp = (float*)d_out;

    char* ws = (char*)d_ws;
    size_t off = 0;
    auto alloc = [&](size_t bytes) -> void* {
        void* p = ws + off;
        off += (bytes + 255) & ~(size_t)255;
        return p;
    };
    unsigned short* c1raw = (unsigned short*)alloc((size_t)NPIX * 32 * 2);
    unsigned short* x1    = (unsigned short*)alloc((size_t)NPIX * 32 * 2);
    unsigned short* c2raw = (unsigned short*)alloc((size_t)NPIX * 64 * 2);
    unsigned short* x2    = (unsigned short*)alloc((size_t)NPIX * 64 * 2);
    unsigned short* c3raw = (unsigned short*)alloc((size_t)NPIX * 256 * 2);
    float* psum  = (float*)alloc((size_t)STATS_NB * 256 * 4);
    float* psq   = (float*)alloc((size_t)STATS_NB * 256 * 4);
    float* aff_a = (float*)alloc(256 * 4);
    float* aff_b = (float*)alloc(256 * 4);
    float* pool8 = (float*)alloc((size_t)BB * 64 * DIMC * 4);
    float* xtok  = (float*)alloc((size_t)BB * TT * DIMC * 4);
    unsigned short* w2c  = (unsigned short*)alloc((size_t)64 * 288 * 2);
    unsigned short* w3c  = (unsigned short*)alloc((size_t)256 * 576 * 2);
    unsigned short* wqkv = (unsigned short*)alloc((size_t)NLD * 1536 * DIMC * 2);
    float* qkvb          = (float*)alloc((size_t)NLD * 1536 * 4);
    unsigned short* wot  = (unsigned short*)alloc((size_t)NLD * 256 * 512 * 2);
    unsigned short* w1t  = (unsigned short*)alloc((size_t)NLD * 1024 * 256 * 2);
    unsigned short* w2t  = (unsigned short*)alloc((size_t)NLD * 256 * 1024 * 2);
    float* qkvbuf = (float*)alloc((size_t)BB * TT * 1536 * 4);
    float* qrel   = (float*)alloc((size_t)BB * NHD * TT * TT * 4);
    float* obuf   = (float*)alloc((size_t)BB * TT * 512 * 4);
    float* hbuf   = (float*)alloc((size_t)BB * TT * DIMC * 4);
    float* mlpbuf = (float*)alloc((size_t)BB * TT * MLPD * 4);
    float* hrow   = (float*)alloc((size_t)BB * DIMC * 4);

    auto cdiv = [](long a, long b) { return (unsigned)((a + b - 1) / b); };

    // --- weight conversions (bf16, [N][K] layouts for WMMA) ---
    k_cvt_convw<<<cdiv(64L * 288, 256), 256, 0, stream>>>(conv2_w, w2c, 64, 32);
    k_cvt_convw<<<cdiv(256L * 576, 256), 256, 0, stream>>>(conv3_w, w3c, 256, 64);
    k_cvt_qkvw<<<cdiv((long)NLD * 1536 * DIMC, 256), 256, 0, stream>>>(Wq, Wk, Wv, wqkv);
    k_cvt_qkvb<<<cdiv((long)NLD * 1536, 256), 256, 0, stream>>>(bq, bk, bv, qkvb);
    k_cvt_t<<<cdiv((long)NLD * 512 * 256, 256), 256, 0, stream>>>(Wo, wot, NLD, 512, 256);
    k_cvt_t<<<cdiv((long)NLD * 256 * 1024, 256), 256, 0, stream>>>(W1, w1t, NLD, 256, 1024);
    k_cvt_t<<<cdiv((long)NLD * 1024 * 256, 256), 256, 0, stream>>>(W2, w2t, NLD, 1024, 256);

    const float invN = 1.f / (float)NPIX;

    // --- conv1 + BN1 + ReLU ---
    k_conv1<<<NPIX * 32 / 256, 256, 0, stream>>>(img, conv1_w, c1raw);
    k_chanstats<<<STATS_NB, 256, 0, stream>>>(c1raw, psum, psq, 32, NPIX);
    k_bnfin<<<1, 256, 0, stream>>>(psum, psq, bn1_g, bn1_b, aff_a, aff_b, 32, invN);
    k_bnrelu<<<cdiv((long)NPIX * 32, 256), 256, 0, stream>>>(c1raw, aff_a, aff_b, x1, 32, (long)NPIX * 32);

    // --- conv2 (WMMA implicit GEMM) + BN2 + ReLU ---
    k_conv_igemm<32, 64, 64><<<dim3(NPIX / 128, 1), 256, 0, stream>>>(x1, w2c, c2raw);
    k_chanstats<<<STATS_NB, 256, 0, stream>>>(c2raw, psum, psq, 64, NPIX);
    k_bnfin<<<1, 256, 0, stream>>>(psum, psq, bn2_g, bn2_b, aff_a, aff_b, 64, invN);
    k_bnrelu<<<cdiv((long)NPIX * 64, 256), 256, 0, stream>>>(c2raw, aff_a, aff_b, x2, 64, (long)NPIX * 64);

    // --- conv3 (WMMA implicit GEMM, 128-channel tile) + BN3 fused into pooling ---
    k_conv_igemm<64, 256, 128><<<dim3(NPIX / 128, 2), 256, 0, stream>>>(x2, w3c, c3raw);
    k_chanstats<<<STATS_NB, 256, 0, stream>>>(c3raw, psum, psq, 256, NPIX);
    k_bnfin<<<1, 256, 0, stream>>>(psum, psq, bn3_g, bn3_b, aff_a, aff_b, 256, invN);
    k_pool8<<<BB * 64, 256, 0, stream>>>(c3raw, aff_a, aff_b, pool8);
    k_tokens<<<cdiv((long)BB * TT * DIMC, 256), 256, 0, stream>>>(pool8, cls, xtok);

    // --- transformer layers ---
    const int M = BB * TT;  // 688
    for (int l = 0; l < NLD; ++l) {
        const unsigned short* wqkv_l = wqkv + (size_t)l * 1536 * DIMC;
        const float* qkvb_l = qkvb + (size_t)l * 1536;
        const float* rel_l = rel + (size_t)l * NHD * TT * DHD;
        const unsigned short* wot_l = wot + (size_t)l * 256 * 512;
        const unsigned short* w1t_l = w1t + (size_t)l * 1024 * 256;
        const unsigned short* w2t_l = w2t + (size_t)l * 256 * 1024;

        k_gemm<<<dim3(cdiv(M, 128), 1536 / 64), 256, 0, stream>>>(
            xtok, wqkv_l, qkvb_l, nullptr, qkvbuf, M, 1536, DIMC);
        k_qrel<<<cdiv((long)BB * NHD * TT * TT, 256), 256, 0, stream>>>(qkvbuf, rel_l, qrel);
        k_attn<<<BB * NHD * TT, 128, 0, stream>>>(qkvbuf, qrel, obuf);
        k_gemm<<<dim3(cdiv(M, 128), 256 / 64), 256, 0, stream>>>(
            obuf, wot_l, bo + (size_t)l * 256, xtok, xtok, M, 256, 512);
        k_ln<<<M, 256, 0, stream>>>(xtok, ln_g + (size_t)l * 256, ln_b + (size_t)l * 256,
                                    hbuf, DIMC, DIMC);
        k_gemm<<<dim3(cdiv(M, 128), 1024 / 64), 256, 0, stream>>>(
            hbuf, w1t_l, b1 + (size_t)l * 1024, nullptr, mlpbuf, M, 1024, DIMC);
        k_gelu<<<cdiv((long)M * MLPD, 256), 256, 0, stream>>>(mlpbuf, (long)M * MLPD);
        k_gemm<<<dim3(cdiv(M, 128), 256 / 64), 256, 0, stream>>>(
            mlpbuf, w2t_l, b2 + (size_t)l * 256, xtok, xtok, M, 256, 1024);
    }

    // --- head: LN + linear on cls token only ---
    k_ln<<<BB, 256, 0, stream>>>(xtok, hln_g, hln_b, hrow, TT * DIMC, DIMC);
    k_head<<<cdiv((long)BB * NOUTD, 256), 256, 0, stream>>>(hrow, Wh, bh, outp);
}
